// RelativeAttention_34488587387517
// MI455X (gfx1250) — matrix operand order repaired and checked by reading, not compiled, not run on hardware
//
#include <hip/hip_runtime.h>

typedef __attribute__((ext_vector_type(16))) _Float16 v16h;
typedef __attribute__((ext_vector_type(8)))  _Float16 v8h;
typedef __attribute__((ext_vector_type(2)))  __fp16   f16x2;   // cvt_pkrtz result type
typedef __attribute__((ext_vector_type(8)))  float    v8f;
typedef __attribute__((ext_vector_type(8)))  int      v8i;

#define MAXREL 8
#define LSEQ   2048
#define DHEAD  64
#define NBH    32                       // B*H
#define LOG2E  1.4426950408889634f
#define QSCALE (0.125f * LOG2E)         // 1/sqrt(64) * log2(e), folded into Q

// lane <-> lane^16 exchange as pure VALU (v_permlanex16_b32, identity sels)
__device__ __forceinline__ int swap16i(int x) {
    return __builtin_amdgcn_permlanex16(x, x, 0x76543210, 0xfedcba98, true, false);
}
__device__ __forceinline__ float swap16f(float x) {
    return __builtin_bit_cast(float, swap16i(__builtin_bit_cast(int, x)));
}

// ---------------------------------------------------------------------------
// Pre-pass: K (f32, [bh][l][d]) -> Kh (f16, same layout)
//           V (f32, [bh][l][d]) -> Vt (f16, [bh][d][l])  (transposed)
// ---------------------------------------------------------------------------
__global__ __launch_bounds__(256) void preconv_kernel(
    const float* __restrict__ K, const float* __restrict__ V,
    _Float16* __restrict__ Kh, _Float16* __restrict__ Vt)
{
    const int bh    = blockIdx.y;
    const int lbase = blockIdx.x * 64;
    const int tid   = threadIdx.x;

    __shared__ _Float16 tile[64][64 + 4];   // [d][l], padded vs bank conflicts

    const float* Kp = K + ((size_t)bh * LSEQ + lbase) * DHEAD;
    const float* Vp = V + ((size_t)bh * LSEQ + lbase) * DHEAD;
    _Float16*   Khp = Kh + ((size_t)bh * LSEQ + lbase) * DHEAD;

#pragma unroll
    for (int i = tid; i < 64 * DHEAD; i += 256)
        Khp[i] = (_Float16)Kp[i];

#pragma unroll
    for (int i = tid; i < 64 * DHEAD; i += 256) {
        int l = i / DHEAD, d = i % DHEAD;
        tile[d][l] = (_Float16)Vp[i];
    }
    __syncthreads();

#pragma unroll
    for (int i = tid; i < 64 * DHEAD; i += 256) {
        int d = i / 64, l = i % 64;
        Vt[((size_t)bh * DHEAD + d) * LSEQ + lbase + l] = tile[d][l];
    }
}

// ---------------------------------------------------------------------------
// Flash attention with relative bias, wave32 + v_wmma_f32_16x16x32_f16.
// S^T = K*Q^T puts the softmax (query) axis on the lane axis of the WMMA C
// layout; O^T = V^T * P^T makes the P conversion an in-lane f16 pack (no exchange).
// K fragments double-buffered across key tiles; V issued early each tile.
// ---------------------------------------------------------------------------
template <bool PRE>
__global__ __launch_bounds__(256) void relattn_fa_kernel(
    const float* __restrict__ Q,  const float* __restrict__ Kf,
    const float* __restrict__ Vf, const _Float16* __restrict__ Kh,
    const _Float16* __restrict__ Vt, const float* __restrict__ bias,
    float* __restrict__ O)
{
    __shared__ float sbias[2 * MAXREL + 1];
    const int tid = threadIdx.x;
    if (tid < 2 * MAXREL + 1) sbias[tid] = bias[tid] * LOG2E;  // log2-domain
    __syncthreads();

    const int lane  = tid & 31;
    const int wave  = tid >> 5;
    const int bh    = blockIdx.y;
    const int qbase = blockIdx.x * 128 + wave * 16;
    const int row16 = lane & 15;
    const bool hi   = lane >= 16;
    const int koff  = hi ? 8 : 0;

    const float b_lo = sbias[0];               // rel <= -8
    const float b_hi = sbias[2 * MAXREL];      // rel >= +8

    const float* Qp = Q + (size_t)bh * LSEQ * DHEAD;
    float*       Op = O + (size_t)bh * LSEQ * DHEAD;

    const _Float16* KhB = PRE ? Kh + ((size_t)bh * LSEQ + row16) * DHEAD + koff : nullptr;
    const _Float16* VtB = PRE ? Vt + ((size_t)bh * DHEAD + row16) * LSEQ + koff : nullptr;
    const float*    KfB = Kf + ((size_t)bh * LSEQ + row16) * DHEAD;
    const float*    VfB = Vf + (size_t)bh * LSEQ * DHEAD + row16;

    auto loadKA = [&](v16h (&kk)[2][2], int kb) {
#pragma unroll
        for (int sub = 0; sub < 2; ++sub) {
            if constexpr (PRE) {
                const _Float16* kr = KhB + (size_t)(kb + sub * 16) * DHEAD;
#pragma unroll
                for (int c = 0; c < 2; ++c) {
                    v8h lo = *(const v8h*)(kr + c * 32);
                    v8h hh = *(const v8h*)(kr + c * 32 + 16);
                    kk[sub][c] = __builtin_shufflevector(lo, hh,
                        0,1,2,3,4,5,6,7,8,9,10,11,12,13,14,15);
                }
            } else {
                const float* krow = KfB + (size_t)(kb + sub * 16) * DHEAD;
#pragma unroll
                for (int c = 0; c < 2; ++c) {
                    const int base = c * 32 + koff;
#pragma unroll
                    for (int e = 0; e < 8; ++e) kk[sub][c][e]     = (_Float16)krow[base + e];
#pragma unroll
                    for (int e = 0; e < 8; ++e) kk[sub][c][8 + e] = (_Float16)krow[base + 16 + e];
                }
            }
        }
    };

    auto loadVA = [&](v16h (&vv)[4], int kb) {
#pragma unroll
        for (int t = 0; t < 4; ++t) {
            if constexpr (PRE) {
                const _Float16* vr = VtB + (size_t)t * 16 * LSEQ + kb;
                v8h lo = *(const v8h*)(vr);
                v8h hh = *(const v8h*)(vr + 16);
                vv[t] = __builtin_shufflevector(lo, hh,
                    0,1,2,3,4,5,6,7,8,9,10,11,12,13,14,15);
            } else {
                const float* vcol = VfB + (size_t)(kb + koff) * DHEAD + t * 16;
#pragma unroll
                for (int e = 0; e < 8; ++e) vv[t][e]     = (_Float16)vcol[(size_t)e * DHEAD];
#pragma unroll
                for (int e = 0; e < 8; ++e) vv[t][8 + e] = (_Float16)vcol[(size_t)(16 + e) * DHEAD];
            }
        }
    };

    // ---- Q^T B-fragments (loop invariant), pre-scaled by 1/sqrt(D)*log2(e).
    const int q = qbase + row16;
    v16h bq[2];
    {
        const float* qrow = Qp + (size_t)q * DHEAD + koff;
#pragma unroll
        for (int c = 0; c < 2; ++c)
#pragma unroll
            for (int e = 0; e < 16; ++e)
                bq[c][e] = (_Float16)(qrow[c * 32 + e + (e & 8)] * QSCALE);
    }

    v8f acc[4];
#pragma unroll
    for (int t = 0; t < 4; ++t) acc[t] = (v8f){0.f,0.f,0.f,0.f,0.f,0.f,0.f,0.f};
    float m_run = -1e30f, l_run = 0.f;

    v16h ka[2][2][2];            // [parity][sub][chunk] — double buffered
    loadKA(ka[0], 0);

#pragma unroll 2
    for (int kt = 0; kt < LSEQ / 32; ++kt) {
        const int kb  = kt * 32;
        const int cur = kt & 1, nxt = cur ^ 1;

        // issue next tile's K fragments + this tile's V fragments up front.
        // PRE path over-reads one tile past the end on the last iteration —
        // lands inside the workspace, harmless; avoids a per-iter branch.
        if (PRE || kt + 1 < LSEQ / 32) loadKA(ka[nxt], kb + 32);
        v16h va[4];
        loadVA(va, kb);

        // ---- S^T = K * Q^T (two 16x16 tiles; K-dim 64 = 2 x 32)
        v8f st[2];
#pragma unroll
        for (int sub = 0; sub < 2; ++sub) {
            v8f c0 = (v8f){0.f,0.f,0.f,0.f,0.f,0.f,0.f,0.f};
            c0 = __builtin_amdgcn_wmma_f32_16x16x32_f16(false, ka[cur][sub][0],
                     false, bq[0], (short)0, c0, false, false);
            c0 = __builtin_amdgcn_wmma_f32_16x16x32_f16(false, ka[cur][sub][1],
                     false, bq[1], (short)0, c0, false, false);
            st[sub] = c0;
        }

        // ---- relative bias: off-diagonal tiles fold one clipped value into
        // the max bookkeeping; only ~2/64 tiles take the gather path.
        float badd;
        if (kb + 31 + MAXREL <= qbase) {
            badd = b_lo;
        } else if (kb >= qbase + 15 + MAXREL) {
            badd = b_hi;
        } else {
            badd = 0.f;
#pragma unroll
            for (int sub = 0; sub < 2; ++sub)
#pragma unroll
                for (int r = 0; r < 8; ++r) {
                    int rel = (kb + sub * 16 + koff + r) - q;
                    rel = rel < -MAXREL ? -MAXREL : (rel > MAXREL ? MAXREL : rel);
                    st[sub][r] += sbias[rel + MAXREL];
                }
        }

        // ---- online softmax (log2 domain). Tree-reduce max (depth 4).
        float mx[8];
#pragma unroll
        for (int r = 0; r < 8; ++r) mx[r] = fmaxf(st[0][r], st[1][r]);
#pragma unroll
        for (int s = 4; s > 0; s >>= 1)
#pragma unroll
            for (int r = 0; r < s; ++r) mx[r] = fmaxf(mx[r], mx[r + s]);
        float mt = mx[0] + badd;
        mt = fmaxf(mt, swap16f(mt));

        float cf = 1.0f;
        if (__ballot(mt > m_run)) {               // rescale only when max moves
            const float m_new = fmaxf(m_run, mt);
            cf = __builtin_amdgcn_exp2f(m_run - m_new);
            m_run = m_new;
#pragma unroll
            for (int t = 0; t < 4; ++t)
#pragma unroll
                for (int r = 0; r < 8; ++r) acc[t][r] *= cf;
        }
        const float msub = m_run - badd;          // exp2(s+badd-m)=exp2(s-msub)

#pragma unroll
        for (int sub = 0; sub < 2; ++sub)
#pragma unroll
            for (int r = 0; r < 8; ++r)
                st[sub][r] = __builtin_amdgcn_exp2f(st[sub][r] - msub);

        float sm[8];
#pragma unroll
        for (int r = 0; r < 8; ++r) sm[r] = st[0][r] + st[1][r];
#pragma unroll
        for (int s = 4; s > 0; s >>= 1)
#pragma unroll
            for (int r = 0; r < s; ++r) sm[r] += sm[r + s];
        float ls = sm[0] + swap16f(sm[0]);
        l_run = l_run * cf + ls;

        // ---- P^T as B-fragment at dword granularity:
        // VGPR j of pb = keys (2j,2j+1) [lanes 0-15] / (8+2j,8+2j+1) [16-31]; VGPR 4+j = keys +16
        v8i pbi;
#pragma unroll
        for (int j = 0; j < 4; ++j) {
            int p0 = __builtin_bit_cast(int,
                         __builtin_amdgcn_cvt_pkrtz(st[0][2*j], st[0][2*j+1]));
            int p1 = __builtin_bit_cast(int,
                         __builtin_amdgcn_cvt_pkrtz(st[1][2*j], st[1][2*j+1]));
            pbi[j]     = p0;
            pbi[4 + j] = p1;
        }
        v16h pb = __builtin_bit_cast(v16h, pbi);

        // ---- O^T += V^T * P^T
#pragma unroll
        for (int t = 0; t < 4; ++t)
            acc[t] = __builtin_amdgcn_wmma_f32_16x16x32_f16(false, va[t],
                         false, pb, (short)0, acc[t], false, false);
    }

    // ---- normalize and store
    const float inv_l = 1.0f / l_run;
    float* orow = Op + (size_t)q * DHEAD + (hi ? 8 : 0);
#pragma unroll
    for (int t = 0; t < 4; ++t)
#pragma unroll
        for (int r = 0; r < 8; ++r)
            orow[t * 16 + r] = acc[t][r] * inv_l;
}

extern "C" void kernel_launch(void* const* d_in, const int* in_sizes, int n_in,
                              void* d_out, int out_size, void* d_ws, size_t ws_size,
                              hipStream_t stream) {
    const float* Q    = (const float*)d_in[0];
    const float* K    = (const float*)d_in[1];
    const float* V    = (const float*)d_in[2];
    const float* bias = (const float*)d_in[3];
    float* O = (float*)d_out;

    dim3 grid(LSEQ / 128, NBH);
    dim3 block(256);

    const size_t elems = (size_t)NBH * LSEQ * DHEAD;           // 4.19M
    const size_t need  = (elems * 2 + 2048) * sizeof(_Float16); // Kh + Vt + overread pad

    if (d_ws != nullptr && ws_size >= need) {
        _Float16* Kh = (_Float16*)d_ws;
        _Float16* Vt = Kh + elems;
        preconv_kernel<<<dim3(LSEQ / 64, NBH), 256, 0, stream>>>(K, V, Kh, Vt);
        relattn_fa_kernel<true><<<grid, block, 0, stream>>>(Q, K, V, Kh, Vt, bias, O);
    } else {
        relattn_fa_kernel<false><<<grid, block, 0, stream>>>(Q, K, V, nullptr, nullptr, bias, O);
    }
}
